// WZMAUltimate_34660386079081
// MI455X (gfx1250) — matrix operand order repaired and checked by reading, not compiled, not run on hardware
//
#include <hip/hip_runtime.h>
#include <hip/hip_bf16.h>

// ---------------------------------------------------------------------------
// MI455X (gfx1250). wave32, WMMA bf16 16x16x32 (f32 accum), async-to-LDS
// staging, all fragment loads vectorized as b128.
// Model: V=8192 D=384 L=6 H=6 HKV=2 HD=64 N=16 B=4 S=1024 DFF=1536
// ---------------------------------------------------------------------------

typedef __bf16 bf16;
typedef __attribute__((ext_vector_type(16))) __bf16 v16bf;
typedef __attribute__((ext_vector_type(8)))  __bf16 v8bf;
typedef __attribute__((ext_vector_type(8)))  float  v8f;

#define D_MODEL 384
#define N_STATE 16
#define N_LAYER 6
#define N_HEAD  6
#define N_KV    2
#define NREP    3
#define HEAD_DIM 64
#define KV_DIM  (N_KV * HEAD_DIM)   // 128
#define SEQ     1024
#define BATCH   4
#define DFF_DIM 1536
#define NTOK    (BATCH * SEQ)       // 4096

__device__ __forceinline__ v8bf ld8(const bf16* p) {
  return *(const v8bf*)p;           // 16B aligned by construction
}
__device__ __forceinline__ v16bf cat8(v8bf lo, v8bf hi) {
  return __builtin_shufflevector(lo, hi, 0, 1, 2, 3, 4, 5, 6, 7, 8, 9, 10, 11,
                                 12, 13, 14, 15);
}
__device__ __forceinline__ v8f wmma_bf16(v16bf a, v16bf b, v8f c) {
  return __builtin_amdgcn_wmma_f32_16x16x32_bf16(false, a, false, b,
                                                 (short)0, c, false, false);
}
// A-layout (16x32 bf16): lane-half h, elems 0..7 -> K = 8h+e ; 8..15 -> 16+8h+e
__device__ __forceinline__ v16bf load_afrag(const bf16* row, int k0, int half) {
  return cat8(ld8(row + k0 + 8 * half), ld8(row + k0 + 16 + 8 * half));
}
// B-layout (32x16 bf16): lane-half h, elems 0..7 -> K = 8h+e ; 8..15 -> 16+8h+e
__device__ __forceinline__ v16bf load_bfrag(const bf16* row, int k0, int half) {
  return cat8(ld8(row + k0 + 8 * half), ld8(row + k0 + 16 + 8 * half));
}

// CDNA5 async global->LDS staging (ASYNCcnt-tracked)
__device__ __forceinline__ void async_ld_b128(void* lds_ptr, const bf16* g) {
  unsigned loff = (unsigned)(size_t)lds_ptr;
  asm volatile("global_load_async_to_lds_b128 %0, %1, off"
               :: "v"(loff), "v"(g) : "memory");
}
__device__ __forceinline__ void wait_async0() {
  asm volatile("s_wait_asynccnt 0" ::: "memory");
}

// ---------------------------------------------------------------------------
// f32 -> bf16 (plain) and f32[K,N] -> bf16[N,K] (transposed, once per launch)
// ---------------------------------------------------------------------------
__global__ __launch_bounds__(256)
void cvt_kernel(const float* __restrict__ in, bf16* __restrict__ out, int n) {
  int i = blockIdx.x * 256 + threadIdx.x;
  if (i < n) out[i] = (bf16)in[i];
}
__global__ __launch_bounds__(256)
void cvt_t_kernel(const float* __restrict__ in, bf16* __restrict__ out,
                  int K, int N) {
  int i = blockIdx.x * 256 + threadIdx.x;
  if (i >= K * N) return;
  int k = i / N, n = i - k * N;
  out[(size_t)n * K + k] = (bf16)in[i];
}

// ---------------------------------------------------------------------------
// Embedding gather (f32 + bf16 copies)
// ---------------------------------------------------------------------------
__global__ __launch_bounds__(D_MODEL)
void gather_kernel(const int* __restrict__ ids, const float* __restrict__ emb,
                   float* __restrict__ x, bf16* __restrict__ xb) {
  int t = blockIdx.x;
  int d = threadIdx.x;
  float v = emb[(size_t)ids[t] * D_MODEL + d];
  x [(size_t)t * D_MODEL + d] = v;
  xb[(size_t)t * D_MODEL + d] = (bf16)v;
}

// ---------------------------------------------------------------------------
// LayerNorm (D=384), one wave per token; optional f32/bf16 outputs
// ---------------------------------------------------------------------------
__global__ __launch_bounds__(256)
void ln_kernel(const float* __restrict__ x, const float* __restrict__ g,
               const float* __restrict__ bt, float* __restrict__ outF,
               bf16* __restrict__ outB, int ntok) {
  int w = threadIdx.x >> 5, lane = threadIdx.x & 31;
  int t = blockIdx.x * 8 + w;
  if (t >= ntok) return;
  const float* row = x + (size_t)t * D_MODEL;
  float v[12];
  float s = 0.f;
#pragma unroll
  for (int i = 0; i < 12; ++i) { v[i] = row[lane + i * 32]; s += v[i]; }
#pragma unroll
  for (int off = 16; off >= 1; off >>= 1) s += __shfl_xor(s, off, 32);
  float mean = s * (1.f / D_MODEL);
  float q = 0.f;
#pragma unroll
  for (int i = 0; i < 12; ++i) { float d = v[i] - mean; q += d * d; }
#pragma unroll
  for (int off = 16; off >= 1; off >>= 1) q += __shfl_xor(q, off, 32);
  float rstd = rsqrtf(q * (1.f / D_MODEL) + 1e-5f);
#pragma unroll
  for (int i = 0; i < 12; ++i) {
    int c = lane + i * 32;
    float o = (v[i] - mean) * rstd * g[c] + bt[c];
    if (outF) outF[(size_t)t * D_MODEL + c] = o;
    if (outB) outB[(size_t)t * D_MODEL + c] = (bf16)o;
  }
}

// ---------------------------------------------------------------------------
// S4 scan: sequential in S, one thread per (b,d) channel; residual fused
// ---------------------------------------------------------------------------
__global__ __launch_bounds__(256)
void s4_kernel(const float* __restrict__ xln, float* __restrict__ x,
               const float* __restrict__ A_log, const float* __restrict__ sB,
               const float* __restrict__ sC, const float* __restrict__ sD) {
  int ch = blockIdx.x * 256 + threadIdx.x;
  if (ch >= BATCH * D_MODEL) return;
  int b = ch / D_MODEL, d = ch % D_MODEL;
  float decay[N_STATE], Bv[N_STATE], Cv[N_STATE], h[N_STATE];
#pragma unroll
  for (int n = 0; n < N_STATE; ++n) {
    decay[n] = 1.f - __expf(A_log[d * N_STATE + n]);
    Bv[n] = sB[d * N_STATE + n];
    Cv[n] = sC[d * N_STATE + n];
    h[n] = 0.f;
  }
  float Dv = sD[d];
  const float* xp = xln + (size_t)b * SEQ * D_MODEL + d;
  float* op = x + (size_t)b * SEQ * D_MODEL + d;
  for (int t = 0; t < SEQ; ++t) {
    float u = xp[(size_t)t * D_MODEL];
    float y = u * Dv;
#pragma unroll
    for (int n = 0; n < N_STATE; ++n) {
      h[n] = h[n] * decay[n] + u * Bv[n];
      y += h[n] * Cv[n];
    }
    op[(size_t)t * D_MODEL] += y;
  }
}

// ---------------------------------------------------------------------------
// GEMM: C[M,N] = act(A[M,K] @ B[K,N] + bias) + resid, with B pre-transposed
// (BT[N][K], row stride = K). One 16x32 output slab per wave (2 WMMA/k-step,
// A fragment reused). 8 waves/block => block covers 16 x 256.
// act: 0=none, 1=tanh, 2=exact gelu.
// ---------------------------------------------------------------------------
__global__ __launch_bounds__(256)
void gemm_kernel(const bf16* __restrict__ A, int lda,
                 const bf16* __restrict__ BT,
                 float* __restrict__ Cf, bf16* __restrict__ Cb, int ldc,
                 const float* __restrict__ bias,
                 const float* __restrict__ resid, int act,
                 int M, int N, int K) {
  int lane = threadIdx.x & 31, w = threadIdx.x >> 5;
  int half = lane >> 4, lm = lane & 15;
  int nt = (blockIdx.x * 8 + w) * 32;
  int mt = blockIdx.y * 16;
  if (nt >= N) return;

  const bf16* Ab  = A  + (size_t)(mt + lm) * lda;
  const bf16* Bt0 = BT + (size_t)(nt + lm) * K;
  const bf16* Bt1 = BT + (size_t)(nt + 16 + lm) * K;

  v8f acc0 = {0, 0, 0, 0, 0, 0, 0, 0};
  v8f acc1 = {0, 0, 0, 0, 0, 0, 0, 0};
  for (int k0 = 0; k0 < K; k0 += 32) {
    v16bf a  = load_afrag(Ab,  k0, half);
    v16bf b0 = load_bfrag(Bt0, k0, half);
    v16bf b1 = load_bfrag(Bt1, k0, half);
    acc0 = wmma_bf16(a, b0, acc0);
    acc1 = wmma_bf16(a, b1, acc1);
  }

  int col0 = nt + lm, col1 = nt + 16 + lm;
  float bv0 = bias ? bias[col0] : 0.f;
  float bv1 = bias ? bias[col1] : 0.f;
#pragma unroll
  for (int r = 0; r < 8; ++r) {
    int row = mt + r + 8 * half;
    float v0 = acc0[r] + bv0;
    float v1 = acc1[r] + bv1;
    if (act == 1) {
      v0 = tanhf(v0); v1 = tanhf(v1);
    } else if (act == 2) {
      v0 = 0.5f * v0 * (1.f + erff(v0 * 0.70710678118f));
      v1 = 0.5f * v1 * (1.f + erff(v1 * 0.70710678118f));
    }
    if (resid) {
      v0 += resid[(size_t)row * ldc + col0];
      v1 += resid[(size_t)row * ldc + col1];
    }
    if (Cf) {
      Cf[(size_t)row * ldc + col0] = v0;
      Cf[(size_t)row * ldc + col1] = v1;
    }
    if (Cb) {
      Cb[(size_t)row * ldc + col0] = (bf16)v0;
      Cb[(size_t)row * ldc + col1] = (bf16)v1;
    }
  }
}

// ---------------------------------------------------------------------------
// V transpose for attention: vb[b*S+s][c] -> vT[b*128 + c][s]
// ---------------------------------------------------------------------------
__global__ __launch_bounds__(KV_DIM)
void vtrans_kernel(const bf16* __restrict__ vb, bf16* __restrict__ vT) {
  int t = blockIdx.x, c = threadIdx.x;
  int b = t >> 10, s = t & (SEQ - 1);
  vT[((size_t)(b * KV_DIM + c)) * SEQ + s] = vb[(size_t)t * KV_DIM + c];
}

// ---------------------------------------------------------------------------
// Flash attention (GQA). 8 waves/block, one 16-query tile per wave; all
// waves of a block share (b, h) so each 32-key K tile is staged once per
// block into LDS via async global->LDS (ASYNCcnt), then fragments are b128
// LDS reads. V comes from the pre-transposed vT with contiguous b128 reads.
// ---------------------------------------------------------------------------
__global__ __launch_bounds__(256)
void attn_kernel(const bf16* __restrict__ q, const bf16* __restrict__ k,
                 const bf16* __restrict__ vT, bf16* __restrict__ out) {
  const int tid  = threadIdx.x;
  const int lane = tid & 31;
  const int w    = tid >> 5;
  const int half = lane >> 4;
  const int lm   = lane & 15;
  const int qt   = blockIdx.x * 8 + w;
  const int h    = blockIdx.y;
  const int b    = blockIdx.z;
  const int kvh  = h / NREP;

  __shared__ __attribute__((aligned(16))) bf16 ldsK[32][HEAD_DIM]; // 4KB
  __shared__ __attribute__((aligned(16))) bf16 ldsP[8][16 * 32];   // 8KB

  // Q tile 16x64 (A layout), fixed for the whole kv loop
  const bf16* qp = q + (size_t)(b * SEQ + qt * 16 + lm) * D_MODEL + h * HEAD_DIM;
  v16bf qa0 = load_afrag(qp, 0, half);
  v16bf qa1 = load_afrag(qp, 32, half);

  v8f o[4];
#pragma unroll
  for (int f = 0; f < 4; ++f) o[f] = (v8f){0, 0, 0, 0, 0, 0, 0, 0};
  float mrow[8], lrow[8];
#pragma unroll
  for (int r = 0; r < 8; ++r) { mrow[r] = -3.0e38f; lrow[r] = 0.f; }

  const float scale = 0.125f;  // 1/sqrt(64)
  const int key0_seg = tid >> 3, seg = tid & 7;  // staging assignment

  for (int kt = 0; kt < SEQ / 32; ++kt) {
    // ---- stage K tile [32 keys x 64] via async global->LDS ----
    {
      const bf16* g = k + (size_t)(b * SEQ + kt * 32 + key0_seg) * KV_DIM +
                      kvh * HEAD_DIM + seg * 8;
      async_ld_b128(&ldsK[key0_seg][seg * 8], g);
      wait_async0();
    }
    __syncthreads();

    // ---- scores: S(16x32) = Q(16x64) . K^T(64x32), 4 WMMA ----
    v8f sc0, sc1;
    {
      v8f zz = {0, 0, 0, 0, 0, 0, 0, 0};
      const bf16* kr0 = &ldsK[lm][0];
      const bf16* kr1 = &ldsK[16 + lm][0];
      v16bf k0a = load_bfrag(kr0, 0, half);
      v16bf k0b = load_bfrag(kr0, 32, half);
      v16bf k1a = load_bfrag(kr1, 0, half);
      v16bf k1b = load_bfrag(kr1, 32, half);
      sc0 = wmma_bf16(qa0, k0a, zz);
      sc0 = wmma_bf16(qa1, k0b, sc0);
      sc1 = wmma_bf16(qa0, k1a, zz);
      sc1 = wmma_bf16(qa1, k1b, sc1);
    }

    // ---- online softmax (row values live across 16 lanes per half) ----
    float corr[8], p0[8], p1[8];
#pragma unroll
    for (int r = 0; r < 8; ++r) {
      float s0 = sc0[r] * scale, s1 = sc1[r] * scale;
      float mx = fmaxf(s0, s1);
#pragma unroll
      for (int off = 8; off >= 1; off >>= 1)
        mx = fmaxf(mx, __shfl_xor(mx, off, 16));
      float mnew = fmaxf(mrow[r], mx);
      corr[r] = __expf(mrow[r] - mnew);
      mrow[r] = mnew;
      p0[r] = __expf(s0 - mnew);
      p1[r] = __expf(s1 - mnew);
      float rs = p0[r] + p1[r];
#pragma unroll
      for (int off = 8; off >= 1; off >>= 1) rs += __shfl_xor(rs, off, 16);
      lrow[r] = lrow[r] * corr[r] + rs;
    }
#pragma unroll
    for (int r = 0; r < 8; ++r) {
      o[0][r] *= corr[r]; o[1][r] *= corr[r];
      o[2][r] *= corr[r]; o[3][r] *= corr[r];
    }

    // ---- P: C layout -> LDS -> A layout (b128 reload) ----
#pragma unroll
    for (int r = 0; r < 8; ++r) {
      int row = r + 8 * half;
      ldsP[w][row * 32 + lm]      = (bf16)p0[r];
      ldsP[w][row * 32 + 16 + lm] = (bf16)p1[r];
    }
    __syncthreads();
    v16bf pa = cat8(ld8(&ldsP[w][lm * 32 + 8 * half]),
                    ld8(&ldsP[w][lm * 32 + 16 + 8 * half]));
    __syncthreads();

    // ---- O += P(16x32) . V(32x64): contiguous b128 from vT ----
#pragma unroll
    for (int f = 0; f < 4; ++f) {
      const bf16* vrow =
          vT + (size_t)(b * KV_DIM + kvh * HEAD_DIM + f * 16 + lm) * SEQ +
          kt * 32;
      v16bf vf = load_bfrag(vrow, 0, half);
      o[f] = wmma_bf16(pa, vf, o[f]);
    }
  }

  // ---- normalize + write [b, s, h*64 + n] ----
  bf16* ob = out + (size_t)(b * SEQ + qt * 16) * D_MODEL + h * HEAD_DIM + lm;
#pragma unroll
  for (int r = 0; r < 8; ++r) {
    int row = r + 8 * half;
    float inv = 1.f / lrow[r];
    ob[(size_t)row * D_MODEL + 0]  = (bf16)(o[0][r] * inv);
    ob[(size_t)row * D_MODEL + 16] = (bf16)(o[1][r] * inv);
    ob[(size_t)row * D_MODEL + 32] = (bf16)(o[2][r] * inv);
    ob[(size_t)row * D_MODEL + 48] = (bf16)(o[3][r] * inv);
  }
}

// ---------------------------------------------------------------------------
// Host orchestration
// ---------------------------------------------------------------------------
static void launch_gemm(hipStream_t s, const bf16* A, int lda, const bf16* BT,
                        float* Cf, bf16* Cb, int ldc, const float* bias,
                        const float* resid, int act, int M, int N, int K) {
  dim3 grid((N + 255) / 256, M / 16, 1);
  gemm_kernel<<<grid, 256, 0, s>>>(A, lda, BT, Cf, Cb, ldc, bias, resid, act,
                                   M, N, K);
}

extern "C" void kernel_launch(void* const* d_in, const int* in_sizes, int n_in,
                              void* d_out, int out_size, void* d_ws,
                              size_t ws_size, hipStream_t stream) {
  (void)in_sizes; (void)n_in; (void)out_size; (void)ws_size;

  const int*   ids    = (const int*)  d_in[0];
  const float* emb    = (const float*)d_in[1];
  const float* hyperW = (const float*)d_in[2];
  const float* A_log  = (const float*)d_in[3];
  const float* s4B    = (const float*)d_in[4];
  const float* s4C    = (const float*)d_in[5];
  const float* s4D    = (const float*)d_in[6];
  const float* Wq     = (const float*)d_in[7];
  const float* Wk     = (const float*)d_in[8];
  const float* Wv     = (const float*)d_in[9];
  const float* Wo     = (const float*)d_in[10];
  const float* bo     = (const float*)d_in[11];
  const float* Wup    = (const float*)d_in[12];
  const float* Wdn    = (const float*)d_in[13];
  const float* bdn    = (const float*)d_in[14];
  const float* g1     = (const float*)d_in[15];
  const float* b1     = (const float*)d_in[16];
  const float* g2     = (const float*)d_in[17];
  const float* b2     = (const float*)d_in[18];
  const float* g3     = (const float*)d_in[19];
  const float* b3     = (const float*)d_in[20];
  const float* gf     = (const float*)d_in[21];
  const float* bf_    = (const float*)d_in[22];

  char* p = (char*)d_ws;
  auto alloc = [&](size_t bytes) -> void* {
    void* r = (void*)p;
    p += (bytes + 255) & ~(size_t)255;
    return r;
  };

  const size_t DD  = (size_t)D_MODEL * D_MODEL;      // 147456
  const size_t DKV = (size_t)D_MODEL * KV_DIM;       // 49152
  const size_t DUP = (size_t)D_MODEL * DFF_DIM;      // 589824

  // transposed bf16 weights: wXT[l] is [N][K]
  bf16* whypT = (bf16*)alloc(DD * 2);
  bf16* wqT   = (bf16*)alloc(N_LAYER * DD * 2);
  bf16* wkT   = (bf16*)alloc(N_LAYER * DKV * 2);
  bf16* wvT   = (bf16*)alloc(N_LAYER * DKV * 2);
  bf16* woT   = (bf16*)alloc(N_LAYER * DD * 2);
  bf16* wupT  = (bf16*)alloc(N_LAYER * DUP * 2);
  bf16* wdnT  = (bf16*)alloc(N_LAYER * DUP * 2);
  float* x    = (float*)alloc((size_t)NTOK * D_MODEL * 4);
  bf16* xb    = (bf16*) alloc((size_t)NTOK * D_MODEL * 2);
  float* xln  = (float*)alloc((size_t)NTOK * D_MODEL * 4);
  bf16* xlnb  = (bf16*) alloc((size_t)NTOK * D_MODEL * 2);
  bf16* qb    = (bf16*) alloc((size_t)NTOK * D_MODEL * 2);
  bf16* kb    = (bf16*) alloc((size_t)NTOK * KV_DIM * 2);
  bf16* vb    = (bf16*) alloc((size_t)NTOK * KV_DIM * 2);
  bf16* vT    = (bf16*) alloc((size_t)NTOK * KV_DIM * 2);
  bf16* attb  = (bf16*) alloc((size_t)NTOK * D_MODEL * 2);
  bf16* upb   = (bf16*) alloc((size_t)NTOK * DFF_DIM * 2);

  auto cvtT = [&](const float* src, bf16* dst, int K, int N) {
    size_t n = (size_t)K * N;
    cvt_t_kernel<<<dim3((unsigned)((n + 255) / 256)), 256, 0, stream>>>(
        src, dst, K, N);
  };
  cvtT(hyperW, whypT, D_MODEL, D_MODEL);
  for (int l = 0; l < N_LAYER; ++l) {
    cvtT(Wq  + l * DD,  wqT  + l * DD,  D_MODEL, D_MODEL);
    cvtT(Wk  + l * DKV, wkT  + l * DKV, D_MODEL, KV_DIM);
    cvtT(Wv  + l * DKV, wvT  + l * DKV, D_MODEL, KV_DIM);
    cvtT(Wo  + l * DD,  woT  + l * DD,  D_MODEL, D_MODEL);
    cvtT(Wup + l * DUP, wupT + l * DUP, D_MODEL, DFF_DIM);
    cvtT(Wdn + l * DUP, wdnT + l * DUP, DFF_DIM, D_MODEL);
  }

  // embedding + hyperdim: x = x + tanh(x @ W)
  gather_kernel<<<NTOK, D_MODEL, 0, stream>>>(ids, emb, x, xb);
  launch_gemm(stream, xb, D_MODEL, whypT, x, nullptr, D_MODEL, nullptr, x,
              /*tanh*/ 1, NTOK, D_MODEL, D_MODEL);

  for (int l = 0; l < N_LAYER; ++l) {
    // S4 branch
    ln_kernel<<<512, 256, 0, stream>>>(x, g1 + l * D_MODEL, b1 + l * D_MODEL,
                                       xln, xlnb, NTOK);
    s4_kernel<<<6, 256, 0, stream>>>(xln, x,
                                     A_log + (size_t)l * D_MODEL * N_STATE,
                                     s4B + (size_t)l * D_MODEL * N_STATE,
                                     s4C + (size_t)l * D_MODEL * N_STATE,
                                     s4D + (size_t)l * D_MODEL);
    // attention branch
    ln_kernel<<<512, 256, 0, stream>>>(x, g2 + l * D_MODEL, b2 + l * D_MODEL,
                                       nullptr, xlnb, NTOK);
    launch_gemm(stream, xlnb, D_MODEL, wqT + l * DD, nullptr, qb, D_MODEL,
                nullptr, nullptr, 0, NTOK, D_MODEL, D_MODEL);
    launch_gemm(stream, xlnb, D_MODEL, wkT + l * DKV, nullptr, kb, KV_DIM,
                nullptr, nullptr, 0, NTOK, KV_DIM, D_MODEL);
    launch_gemm(stream, xlnb, D_MODEL, wvT + l * DKV, nullptr, vb, KV_DIM,
                nullptr, nullptr, 0, NTOK, KV_DIM, D_MODEL);
    vtrans_kernel<<<NTOK, KV_DIM, 0, stream>>>(vb, vT);
    attn_kernel<<<dim3(SEQ / 16 / 8, N_HEAD, BATCH), 256, 0, stream>>>(
        qb, kb, vT, attb);
    launch_gemm(stream, attb, D_MODEL, woT + l * DD, x, nullptr, D_MODEL,
                bo + l * D_MODEL, x, 0, NTOK, D_MODEL, D_MODEL);
    // MLP branch
    ln_kernel<<<512, 256, 0, stream>>>(x, g3 + l * D_MODEL, b3 + l * D_MODEL,
                                       nullptr, xlnb, NTOK);
    launch_gemm(stream, xlnb, D_MODEL, wupT + l * DUP, nullptr, upb, DFF_DIM,
                nullptr, nullptr, /*gelu*/ 2, NTOK, DFF_DIM, D_MODEL);
    launch_gemm(stream, upb, DFF_DIM, wdnT + l * DUP, x, nullptr, D_MODEL,
                bdn + l * D_MODEL, x, 0, NTOK, D_MODEL, DFF_DIM);
  }

  // final LN -> f32 output
  ln_kernel<<<512, 256, 0, stream>>>(x, gf, bf_, (float*)d_out, nullptr, NTOK);
}
